// DiffusionReturnPrediction_10368051052858
// MI455X (gfx1250) — matrix operand order repaired and checked by reading, not compiled, not run on hardware
//
#include <hip/hip_runtime.h>
#include <hip/hip_bf16.h>

typedef __bf16 bf16;
typedef __attribute__((ext_vector_type(16))) __bf16 v16bf;
typedef __attribute__((ext_vector_type(8)))  float  v8f;

union Frag16 { v16bf v; uint4 q[2]; };

#define BB 1024
#define TT 180
#define NN 14
#define FF 4
#define DD 10080          // N*T*F
#define DDP 10240         // padded to multiple of 256
#define SNH 1024
#define HH 128
#define BN 14336          // B*N
#define NOUT 8

// ---------------- helpers ----------------
__device__ inline v8f wmma_bf16(v16bf a, v16bf b, v8f c) {
    return __builtin_amdgcn_wmma_f32_16x16x32_bf16(false, a, false, b, (short)0, c, false, false);
}
__device__ inline float sigmoidf_(float v) { return 1.0f / (1.0f + __expf(-v)); }

// A fragment (16x32 bf16, M x K): lane m=L&15, two 16B chunks at k+hi*8 / k+16+hi*8
__device__ inline v16bf ldfragA(const bf16* p) {
    Frag16 f;
    f.q[0] = *(const uint4*)p;
    f.q[1] = *(const uint4*)(p + 16);
    return f.v;
}
// B fragment (32x16 bf16, K x N): lane n=L&15, k rows hi*8+s (s<8) / 16+hi*8+(s-8), stride N
__device__ inline v16bf ldfragB(const bf16* p, unsigned N) {
    v16bf v;
#pragma unroll
    for (int s = 0; s < 16; ++s) v[s] = p[(unsigned)(s < 8 ? s : s + 8) * N];
    return v;
}

// ---------------- prep kernels ----------------
__global__ void k_f2b(const float* __restrict__ s, bf16* __restrict__ d, int n) {
    int i = blockIdx.x * blockDim.x + threadIdx.x;
    if (i < n) d[i] = (bf16)s[i];
}

// convert with zero-padded columns: src [rows][ncol] -> dst [rows][npad]
__global__ void k_f2b_pad(const float* __restrict__ s, bf16* __restrict__ d,
                          int rows, int ncol, int npad) {
    int i = blockIdx.x * blockDim.x + threadIdx.x;
    if (i >= rows * npad) return;
    int n = i % npad, r = i / npad;
    d[i] = (n < ncol) ? (bf16)s[r * ncol + n] : (bf16)0.0f;
}

// x [B,T,N,F] -> Xflat bf16 [B, (n*T+t)*F+f]  and  Xcomb[t][b*14+n][0..3]
__global__ void k_xflat(const float* __restrict__ x, bf16* __restrict__ xf, bf16* __restrict__ xcomb) {
    int i = blockIdx.x * blockDim.x + threadIdx.x;
    if (i >= BB * TT * NN * FF) return;
    int f  = i & 3;
    int i2 = i >> 2;
    int n  = i2 % NN;
    int i3 = i2 / NN;
    int t  = i3 % TT;
    int b  = i3 / TT;
    float v = x[i];
    xf[(size_t)b * DD + (n * TT + t) * FF + f] = (bf16)v;
    xcomb[((size_t)t * BN + (size_t)b * NN + n) * 8 + f] = (bf16)v;
}

// Augmented LSTM weight [K=160][N=512] pre-swizzled into B-fragment order:
// waugF[((kb*32+nt)*32+L)*16 + j] = Waug[kb*32 + (j<8 ? 8*(L>>4)+j : 16+8*(L>>4)+j-8)][nt*16+(L&15)]
// rows 0..7 = Wih^T, 8..31 zero pad, 32..159 = Whh^T. Also biasC = bih+bhh.
__global__ void k_waug(const float* __restrict__ Wih, const float* __restrict__ Whh,
                       const float* __restrict__ bih, const float* __restrict__ bhh,
                       bf16* __restrict__ waugF, float* __restrict__ biasC) {
    int i = blockIdx.x * blockDim.x + threadIdx.x;
    if (i < 5 * 32 * 32 * 16) {
        int j  = i & 15;
        int L  = (i >> 4) & 31;
        int nt = (i >> 9) & 31;
        int kb = i >> 14;
        int k  = kb * 32 + ((j < 8) ? (L >> 4) * 8 + j : 16 + (L >> 4) * 8 + (j - 8));
        int n  = nt * 16 + (L & 15);
        float v = 0.0f;
        if (k < 8)        v = Wih[n * 8 + k];
        else if (k >= 32) v = Whh[n * HH + (k - 32)];
        waugF[i] = (bf16)v;
    }
    if (i < 512) biasC[i] = bih[i] + bhh[i];
}

// GCN normalized adjacency (14x14), serial on one lane (56 edges)
__global__ void k_adj(const int* __restrict__ ei, float* __restrict__ A14) {
    if (threadIdx.x == 0 && blockIdx.x == 0) {
        float deg[NN];
        for (int i = 0; i < NN; i++) deg[i] = 1.0f;           // self loop
        for (int k = 0; k < 56; k++) deg[ei[56 + k]] += 1.0f;
        float dinv[NN];
        for (int i = 0; i < NN; i++) dinv[i] = rsqrtf(deg[i]);
        for (int i = 0; i < NN * NN; i++) A14[i] = 0.0f;
        for (int k = 0; k < 56; k++) {
            int r = ei[k], c = ei[56 + k];
            A14[c * NN + r] += dinv[r] * dinv[c];
        }
        for (int i = 0; i < NN; i++) A14[i * NN + i] += dinv[i] * dinv[i];
    }
}

// ---------------- generic WMMA GEMM ----------------
// C = A[MxK] * B[KxN], bf16, N already padded so every tile is in-range.
// Wave computes a 32x64 strip; 8 waves arranged (8>>wnw_shift) x (1<<wnw_shift).
// epi 0: silu(acc + 0.1*ws[n] + b1[n]) -> bf16 H1out[m*N+n]
// epi 1: (acc + b2[n]) -> bf16 scatter into Xcomb (scores half), guarded n<NREAL
// epi 2: acc -> float Cout[m*N+n]
__global__ __launch_bounds__(256)
void gemm_bf16_kernel(const bf16* __restrict__ A, const bf16* __restrict__ B,
                      int M, int N, int K, int wnw_shift, int NREAL, int epi,
                      const float* __restrict__ e_ws, const float* __restrict__ e_b1,
                      bf16* __restrict__ H1out,
                      const float* __restrict__ e_b2, bf16* __restrict__ Xcomb,
                      float* __restrict__ Cout) {
    int lane = threadIdx.x & 31;
    int wave = threadIdx.x >> 5;
    int wnw  = 1 << wnw_shift;
    int wn   = wave & (wnw - 1);
    int wm   = wave >> wnw_shift;
    int m0    = (blockIdx.y * (8 >> wnw_shift) + wm) * 32;
    int nbase = (blockIdx.x * wnw + wn) * 64;
    int ml = lane & 15, hi = lane >> 4;

    const bf16* pa0 = A + (unsigned)((m0 + ml) * K) + hi * 8;
    const bf16* pa1 = pa0 + (unsigned)(16 * K);
    const bf16* pb  = B + (unsigned)(hi * 8 * N) + nbase + ml;
    const unsigned bstep = 32u * (unsigned)N;

    v8f acc[2][4] = {};

    v16bf ca0 = ldfragA(pa0);
    v16bf ca1 = ldfragA(pa1);
    v16bf cb0 = ldfragB(pb, (unsigned)N);
    v16bf cb1 = ldfragB(pb + 16, (unsigned)N);
    v16bf cb2 = ldfragB(pb + 32, (unsigned)N);
    v16bf cb3 = ldfragB(pb + 48, (unsigned)N);

    int steps = K >> 5;
#pragma unroll 2
    for (int s = 0; s < steps - 1; ++s) {
        pa0 += 32; pa1 += 32; pb += bstep;
        v16bf na0 = ldfragA(pa0);
        v16bf na1 = ldfragA(pa1);
        v16bf nb0 = ldfragB(pb, (unsigned)N);
        v16bf nb1 = ldfragB(pb + 16, (unsigned)N);
        v16bf nb2 = ldfragB(pb + 32, (unsigned)N);
        v16bf nb3 = ldfragB(pb + 48, (unsigned)N);
        acc[0][0] = wmma_bf16(ca0, cb0, acc[0][0]);
        acc[0][1] = wmma_bf16(ca0, cb1, acc[0][1]);
        acc[0][2] = wmma_bf16(ca0, cb2, acc[0][2]);
        acc[0][3] = wmma_bf16(ca0, cb3, acc[0][3]);
        acc[1][0] = wmma_bf16(ca1, cb0, acc[1][0]);
        acc[1][1] = wmma_bf16(ca1, cb1, acc[1][1]);
        acc[1][2] = wmma_bf16(ca1, cb2, acc[1][2]);
        acc[1][3] = wmma_bf16(ca1, cb3, acc[1][3]);
        ca0 = na0; ca1 = na1;
        cb0 = nb0; cb1 = nb1; cb2 = nb2; cb3 = nb3;
    }
    acc[0][0] = wmma_bf16(ca0, cb0, acc[0][0]);
    acc[0][1] = wmma_bf16(ca0, cb1, acc[0][1]);
    acc[0][2] = wmma_bf16(ca0, cb2, acc[0][2]);
    acc[0][3] = wmma_bf16(ca0, cb3, acc[0][3]);
    acc[1][0] = wmma_bf16(ca1, cb0, acc[1][0]);
    acc[1][1] = wmma_bf16(ca1, cb1, acc[1][1]);
    acc[1][2] = wmma_bf16(ca1, cb2, acc[1][2]);
    acc[1][3] = wmma_bf16(ca1, cb3, acc[1][3]);

#pragma unroll
    for (int rb = 0; rb < 2; ++rb) {
#pragma unroll
        for (int s4 = 0; s4 < 4; ++s4) {
            int nn = nbase + s4 * 16 + ml;
            if (epi == 0) {
                float wsv = e_ws[nn], b1v = e_b1[nn];
#pragma unroll
                for (int r = 0; r < 8; r++) {
                    int m = m0 + rb * 16 + r + 8 * hi;
                    float v = acc[rb][s4][r] + 0.1f * wsv + b1v;
                    H1out[(size_t)m * N + nn] = (bf16)(v * sigmoidf_(v));
                }
            } else if (epi == 1) {
                if (nn < NREAL) {
                    float b2v = e_b2[nn];
                    int node = nn / 720;
                    int rem  = nn - node * 720;
                    int t = rem >> 2, f = rem & 3;
#pragma unroll
                    for (int r = 0; r < 8; r++) {
                        int m = m0 + rb * 16 + r + 8 * hi;      // batch index
                        float v = acc[rb][s4][r] + b2v;
                        Xcomb[((size_t)t * BN + (size_t)m * NN + node) * 8 + 4 + f] = (bf16)v;
                    }
                }
            } else {
#pragma unroll
                for (int r = 0; r < 8; r++) {
                    int m = m0 + rb * 16 + r + 8 * hi;
                    Cout[(size_t)m * N + nn] = acc[rb][s4][r];
                }
            }
        }
    }
}

// ---------------- LSTM persistent kernel ----------------
// 4 waves/block, wave owns 16 sequences; 180 steps; K augmented to 160.
__global__ __launch_bounds__(128)
void lstm_kernel(const bf16* __restrict__ xcomb,   // [180][BN][8]
                 const bf16* __restrict__ waugF,   // [5][32][32][16]
                 const float* __restrict__ biasC,  // [512]
                 bf16* __restrict__ hT) {          // [BN][128]
    __shared__ __align__(16) bf16  sW[5 * 32 * 32 * 16];   // 160 KB
    __shared__ __align__(16) float sB[512];
    __shared__ __align__(16) bf16  sH[4][16][136];         // per-wave h tile (padded)

    int tid = threadIdx.x;
    int lane = tid & 31, wave = tid >> 5;
    {
        const uint4* src = (const uint4*)waugF;
        uint4* dst = (uint4*)sW;
        for (int i = tid; i < 5 * 32 * 32 * 16 / 8; i += 128) dst[i] = src[i];
        for (int i = tid; i < 512; i += 128) sB[i] = biasC[i];
    }
    __syncthreads();

    int row0 = blockIdx.x * 64 + wave * 16;
    int m = lane & 15, hi = lane >> 4;

    Frag16 zf16; zf16.q[0] = uint4{0, 0, 0, 0}; zf16.q[1] = uint4{0, 0, 0, 0};
    v16bf hA[4];
#pragma unroll
    for (int i = 0; i < 4; i++) hA[i] = zf16.v;
    v8f cst[8] = {};

    for (int t = 0; t < TT; ++t) {
        // input fragment: k 0..7 = x_comb, rest zero-padded
        Frag16 xf;
        xf.q[1] = uint4{0, 0, 0, 0};
        if (hi == 0) xf.q[0] = *(const uint4*)(xcomb + ((size_t)t * BN + row0 + m) * 8);
        else         xf.q[0] = uint4{0, 0, 0, 0};

#pragma unroll
        for (int j = 0; j < 8; ++j) {        // 16-col group within each gate
            v8f zg4[4];
#pragma unroll
            for (int g4 = 0; g4 < 4; ++g4) { // i, f, g, o
                int nt = g4 * 8 + j;
                v8f a = {};
#pragma unroll
                for (int kb = 0; kb < 5; ++kb) {
                    const bf16* bp = &sW[((kb * 32 + nt) * 32 + lane) * 16];
                    Frag16 bf;
                    bf.q[0] = *(const uint4*)bp;
                    bf.q[1] = *(const uint4*)(bp + 8);
                    v16bf av = (kb == 0) ? xf.v : hA[kb - 1];
                    a = wmma_bf16(av, bf.v, a);
                }
                float bb = sB[nt * 16 + m];
#pragma unroll
                for (int r = 0; r < 8; r++) a[r] += bb;
                zg4[g4] = a;
            }
#pragma unroll
            for (int r = 0; r < 8; r++) {
                float iv = sigmoidf_(zg4[0][r]);
                float fv = sigmoidf_(zg4[1][r]);
                float gv = tanhf(zg4[2][r]);
                float ov = sigmoidf_(zg4[3][r]);
                float c = fv * cst[j][r] + iv * gv;
                cst[j][r] = c;
                sH[wave][r + 8 * hi][j * 16 + m] = (bf16)(ov * tanhf(c));
            }
        }
        // rebuild h A-fragments (D-layout -> A-layout via LDS)
#pragma unroll
        for (int hb = 0; hb < 4; ++hb) {
            const bf16* p = &sH[wave][m][hb * 32 + hi * 8];
            Frag16 f;
            f.q[0] = *(const uint4*)p;
            f.q[1] = *(const uint4*)(p + 16);
            hA[hb] = f.v;
        }
    }
    // write final h (16 rows x 128 cols) to global
#pragma unroll
    for (int it = 0; it < 8; ++it) {
        int idx = it * 32 + lane;
        int rr = idx >> 4, ch = idx & 15;
        *(uint4*)(hT + (size_t)(row0 + rr) * HH + ch * 8) = *(const uint4*)(&sH[wave][rr][ch * 8]);
    }
}

// ---------------- tail kernels ----------------
__global__ void k_feats(const float* __restrict__ tmp, const float* __restrict__ A14,
                        const float* __restrict__ gcnb, float* __restrict__ feats) {
    int idx = blockIdx.x * blockDim.x + threadIdx.x;   // B*N*G
    if (idx >= BB * NN * HH) return;
    int g = idx & 127;
    int bi = idx >> 7;
    int i = bi % NN, b = bi / NN;
    float acc = gcnb[g];
#pragma unroll
    for (int j = 0; j < NN; j++)
        acc += A14[i * NN + j] * tmp[((size_t)b * NN + j) * HH + g];
    feats[idx] = acc;
}

__global__ void k_hid(const float* __restrict__ feats, const float* __restrict__ W1,
                      const float* __restrict__ b1, float* __restrict__ hid) {
    int idx = blockIdx.x * blockDim.x + threadIdx.x;   // BN*64
    if (idx >= BN * 64) return;
    int o = idx & 63, row = idx >> 6;
    const float* fr = feats + (size_t)row * HH;
    float acc = b1[o];
    for (int g = 0; g < HH; g++) acc += fr[g] * W1[g * 64 + o];
    hid[idx] = acc * sigmoidf_(acc);
}

__global__ void k_hub(const float* __restrict__ hid, const float* __restrict__ W2,
                      const float* __restrict__ b2, float* __restrict__ hub) {
    int row = blockIdx.x * blockDim.x + threadIdx.x;   // BN
    if (row >= BN) return;
    const float* hr = hid + (size_t)row * 64;
    float acc = b2[0];
    for (int o = 0; o < 64; o++) acc += hr[o] * W2[o];
    hub[row] = acc;
}

__global__ void k_pool(const float* __restrict__ hub, const float* __restrict__ pW,
                       const float* __restrict__ pb, float* __restrict__ out) {
    int idx = blockIdx.x * blockDim.x + threadIdx.x;   // B*NOUT
    if (idx >= BB * NOUT) return;
    int o = idx & 7, b = idx >> 3;
    float acc = pb[o];
#pragma unroll
    for (int n = 0; n < NN; n++) acc += hub[b * NN + n] * pW[n * NOUT + o];
    out[idx] = acc;
}

// ---------------- launch ----------------
extern "C" void kernel_launch(void* const* d_in, const int* in_sizes, int n_in,
                              void* d_out, int out_size, void* d_ws, size_t ws_size,
                              hipStream_t stream) {
    const float* x    = (const float*)d_in[0];
    const int*   ei   = (const int*)d_in[1];
    const float* snW1 = (const float*)d_in[2];
    const float* snws = (const float*)d_in[3];
    const float* snb1 = (const float*)d_in[4];
    const float* snW2 = (const float*)d_in[5];
    const float* snb2 = (const float*)d_in[6];
    const float* Wih  = (const float*)d_in[7];
    const float* Whh  = (const float*)d_in[8];
    const float* bih  = (const float*)d_in[9];
    const float* bhh  = (const float*)d_in[10];
    const float* gcnW = (const float*)d_in[11];
    const float* gcnb = (const float*)d_in[12];
    const float* mW1  = (const float*)d_in[13];
    const float* mb1  = (const float*)d_in[14];
    const float* mW2  = (const float*)d_in[15];
    const float* mb2  = (const float*)d_in[16];
    const float* pW   = (const float*)d_in[17];
    const float* pb   = (const float*)d_in[18];
    float* out = (float*)d_out;

    char* w = (char*)d_ws;
    auto carve = [&](size_t bytes) -> char* {
        char* p = w;
        w += (bytes + 255) & ~(size_t)255;
        return p;
    };
    bf16*  XF    = (bf16*)carve((size_t)BB * DD * 2);        // x flat bf16 [1024][10080]
    bf16*  W1B   = (bf16*)carve((size_t)DD * SNH * 2);       // [10080][1024]
    bf16*  W2B   = (bf16*)carve((size_t)SNH * DDP * 2);      // [1024][10240] zero-padded
    bf16*  H1B   = (bf16*)carve((size_t)BB * SNH * 2);
    bf16*  XCOMB = (bf16*)carve((size_t)TT * BN * 8 * 2);
    bf16*  WAUG  = (bf16*)carve((size_t)5 * 32 * 32 * 16 * 2);
    float* BIASC = (float*)carve(512 * 4);
    bf16*  GCNWB = (bf16*)carve((size_t)HH * HH * 2);
    bf16*  HT    = (bf16*)carve((size_t)BN * HH * 2);
    float* TMP   = (float*)carve((size_t)BN * HH * 4);
    float* A14   = (float*)carve(NN * NN * 4);
    float* FEATS = (float*)carve((size_t)BN * HH * 4);
    float* HID   = (float*)carve((size_t)BN * 64 * 4);
    float* HUB   = (float*)carve((size_t)BN * 4);

    const int TB = 256;
    // ---- prep ----
    k_f2b<<<(DD * SNH + TB - 1) / TB, TB, 0, stream>>>(snW1, W1B, DD * SNH);
    k_f2b_pad<<<(SNH * DDP + TB - 1) / TB, TB, 0, stream>>>(snW2, W2B, SNH, DD, DDP);
    k_f2b<<<(HH * HH + TB - 1) / TB, TB, 0, stream>>>(gcnW, GCNWB, HH * HH);
    k_xflat<<<(BB * TT * NN * FF + TB - 1) / TB, TB, 0, stream>>>(x, XF, XCOMB);
    k_waug<<<(5 * 32 * 32 * 16 + TB - 1) / TB, TB, 0, stream>>>(Wih, Whh, bih, bhh, WAUG, BIASC);
    k_adj<<<1, 32, 0, stream>>>(ei, A14);

    // ---- score net ----
    // L1: [1024 x 10080] @ [10080 x 1024], blocks 64 rows x 256 cols
    gemm_bf16_kernel<<<dim3(SNH / 256, BB / 64), TB, 0, stream>>>(
        XF, W1B, BB, SNH, DD, 2, SNH, 0, snws, snb1, H1B, nullptr, nullptr, nullptr);
    // L2: [1024 x 1024] @ [1024 x 10240(pad)], scatter scores into Xcomb
    gemm_bf16_kernel<<<dim3(DDP / 256, BB / 64), TB, 0, stream>>>(
        H1B, W2B, BB, DDP, SNH, 2, DD, 1, nullptr, nullptr, nullptr, snb2, XCOMB, nullptr);

    // ---- LSTM ----
    lstm_kernel<<<BN / 64, 128, 0, stream>>>(XCOMB, WAUG, BIASC, HT);

    // ---- GCN projection h_t @ gcn_W : blocks 128 rows x 128 cols ----
    gemm_bf16_kernel<<<dim3(1, BN / 128), TB, 0, stream>>>(
        HT, GCNWB, BN, HH, HH, 1, HH, 2, nullptr, nullptr, nullptr, nullptr, nullptr, TMP);

    // ---- tail ----
    k_feats<<<(BB * NN * HH + TB - 1) / TB, TB, 0, stream>>>(TMP, A14, gcnb, FEATS);
    k_hid<<<(BN * 64 + TB - 1) / TB, TB, 0, stream>>>(FEATS, mW1, mb1, HID);
    k_hub<<<(BN + TB - 1) / TB, TB, 0, stream>>>(HID, mW2, mb2, HUB);
    k_pool<<<(BB * NOUT + TB - 1) / TB, TB, 0, stream>>>(HUB, pW, pb, out);
}
